// DKTForgetTotal_54090818126191
// MI455X (gfx1250) — compile-verified, matrix operand order checked by reading, not run on hardware
//
#include <hip/hip_runtime.h>
#include <math.h>

// ---- problem constants (match reference) ----
#define BB   8
#define LL   200
#define MM   4
#define SS   100
#define DD   128
#define FF   4
#define NTOT 300           // 3 * FF * 25
#define NROW (BB*LL)       // 1600
#define LPAD 208           // LL padded to multiple of 16
#define WPRE_STRIDE (DD + NTOT)   // 428 rows of W_pre per skill

typedef __attribute__((ext_vector_type(16))) _Float16 v16h;
typedef __attribute__((ext_vector_type(8)))  _Float16 v8h;
typedef __attribute__((ext_vector_type(2)))  _Float16 v2h;
typedef __attribute__((ext_vector_type(8)))  float    v8f;

__device__ __forceinline__ float sigf(float x) { return 1.0f / (1.0f + expf(-x)); }

// =====================================================================
// Stage 1: attention + concat builder.
// One block (128 threads, thread d owns dim d) per row n.
// xcat[n,384] = [emb_q | emb_r | att]; scat[n,256] = [emb_sq | s_att].
// Softmax over S entries where inactive skills carry score exactly 0.
// =====================================================================
__global__ void attn_concat_kernel(const int* __restrict__ q, const int* __restrict__ c,
                                   const int* __restrict__ sq, const int* __restrict__ sc,
                                   const int* __restrict__ r,
                                   const float* __restrict__ Eq, const float* __restrict__ Es,
                                   const float* __restrict__ Er,
                                   float* __restrict__ xcat, float* __restrict__ scat)
{
    const int n = blockIdx.x;
    const int d = threadIdx.x;            // 0..127
    const int lane = d & 31, wave = d >> 5;
    __shared__ float red[4];
    const float inv_sqrt_d = 0.08838834764831845f;  // 1/sqrt(128)

    for (int variant = 0; variant < 2; ++variant) {
        const int* qq = variant ? sq : q;
        const int* cc = variant ? sc : c;
        float eq = Eq[qq[n] * DD + d];

        int   cm[MM];
        float sco[MM];
        int nvalid = 0;
        for (int m = 0; m < MM; ++m) { cm[m] = cc[n * MM + m]; if (cm[m] >= 0) nvalid++; }

        for (int m = 0; m < MM; ++m) {
            float p = (cm[m] >= 0) ? eq * Es[cm[m] * DD + d] : 0.0f;
            for (int o = 16; o > 0; o >>= 1) p += __shfl_xor(p, o);
            if (lane == 0) red[wave] = p;
            __syncthreads();
            sco[m] = (cm[m] >= 0) ? (red[0] + red[1] + red[2] + red[3]) * inv_sqrt_d : 0.0f;
            __syncthreads();
        }
        float mx = 0.0f;
        for (int m = 0; m < MM; ++m) if (cm[m] >= 0) mx = fmaxf(mx, sco[m]);
        float Z = (float)(SS - nvalid) * expf(-mx);
        float w[MM];
        for (int m = 0; m < MM; ++m) {
            w[m] = (cm[m] >= 0) ? expf(sco[m] - mx) : 0.0f;
            Z += w[m];
        }
        float att = 0.0f;
        for (int m = 0; m < MM; ++m)
            if (cm[m] >= 0) att += (w[m] / Z) * Es[cm[m] * DD + d];

        if (variant == 0) {
            xcat[n * 384 + d]          = eq;
            xcat[n * 384 + DD + d]     = Er[r[n] * DD + d];
            xcat[n * 384 + 2 * DD + d] = att;
        } else {
            scat[n * 256 + d]      = eq;
            scat[n * 256 + DD + d] = att;
        }
        __syncthreads();
    }
}

// =====================================================================
// Stage 2: sparse forget path (exploits the 12-hot structure of fv1).
// fh[n] = b_pre + sum_{m valid} [ (E_skill[c] * (sum_12 W_cemb rows)) @ Wp_theta[c]
//                                 + sum_12 Wp_fv[c] rows ]
// One block (128 threads) per row; matvec staged through LDS; row gathers
// are coalesced (thread d reads column d of each gathered weight row).
// =====================================================================
__global__ void forget_kernel(const int* __restrict__ c, const int* __restrict__ rg,
                              const int* __restrict__ pc, const int* __restrict__ ac,
                              const float* __restrict__ Es, const float* __restrict__ Wc,
                              const float* __restrict__ Wp, const float* __restrict__ bpre,
                              float* __restrict__ out)
{
    const int n = blockIdx.x;
    const int d = threadIdx.x;            // 0..127
    __shared__ float xsh[DD];

    float acc = bpre[d];
    for (int m = 0; m < MM; ++m) {
        const int cm = c[n * MM + m];     // uniform across block
        if (cm >= 0) {
            int idx[12];
            const int rgv = rg[n * MM + m];
            for (int f = 0; f < FF; ++f) {
                idx[f]     =       f * 25 + rgv;
                idx[4 + f] = 100 + f * 25 + pc[(n * MM + m) * FF + f];
                idx[8 + f] = 200 + f * 25 + ac[(n * MM + m) * FF + f];
            }
            float ce = 0.0f;
            const float* wfv = Wp + (cm * WPRE_STRIDE + DD) * DD;   // fv section
            for (int j = 0; j < 12; ++j) {
                ce  += Wc [idx[j] * DD + d];
                acc += wfv[idx[j] * DD + d];
            }
            xsh[d] = Es[cm * DD + d] * ce;
            __syncthreads();
            const float* wth = Wp + cm * WPRE_STRIDE * DD;          // theta section
            float a2 = 0.0f;
            for (int k = 0; k < DD; ++k) a2 += xsh[k] * wth[k * DD + d];
            acc += a2;
            __syncthreads();
        }
    }
    out[n * DD + d] = acc;
}

// =====================================================================
// WMMA GEMM with LDS-staged tiles, no bounds checks (caller pads M to 16,
// N to 64, K to 32; splitK is a multiple of 32).
//   C[M,N] = act( [A0|A1][M,K] * [B0;B1][K,N] + bias0(+bias1) )
// btrans=1: B element (k,n) stored at Bp[n*ldb + k]  (LSTM W^T usage).
// Block = 128 threads = 4 waves; block tile = 16(M) x 64(N); each wave
// computes one 16x16 tile with v_wmma_f32_16x16x32_f16 (f32 accumulate).
//
// LDS layouts: As[m][kk], Bs[n][kk] (B transposed at staging). For a lane
// (hi = lane>>4, mn = lane&15) the 16 operand halves are exactly the two
// contiguous 16-byte runs kk = hi*8..hi*8+7 and 16+hi*8..16+hi*8+7, so each
// operand is read with two ds_load_b128 and a compile-time shufflevector.
// =====================================================================
__global__ void gemm_wmma_kernel(const float* __restrict__ A0, const float* __restrict__ A1,
                                 int lda0, int lda1, int splitK,
                                 const float* __restrict__ B0, const float* __restrict__ B1,
                                 int ldb0, int ldb1, int btrans,
                                 const float* __restrict__ bias0, const float* __restrict__ bias1,
                                 float* __restrict__ C, int ldc, int K, int act)
{
    const int t    = threadIdx.x;
    const int lane = t & 31;
    const int wave = t >> 5;
    const int hi   = lane >> 4;           // 0 or 1
    const int mn   = lane & 15;
    const int mbase = blockIdx.x * 16;
    const int nbase = blockIdx.y * 64;

    __shared__ __align__(16) _Float16 As[16 * 32];   // [m][kk]
    __shared__ __align__(16) _Float16 Bs[64 * 32];   // [n][kk]   (transposed)

    v8f acc = {};
    for (int k0 = 0; k0 < K; k0 += 32) {
        // uniform per-k-step part selection (splitK is a multiple of 32)
        const float* Ap; int la;
        const float* Bp; int lb; int koff;
        if (k0 < splitK) { Ap = A0 + k0;            la = lda0;
                           Bp = B0; lb = ldb0; koff = k0; }
        else             { Ap = A1 + (k0 - splitK); la = lda1;
                           Bp = B1; lb = ldb1; koff = k0 - splitK; }

        // ---- phase 1: batched global loads into registers (float2, 8B) ----
        float2 ta[2];
#pragma unroll
        for (int i = 0; i < 2; ++i) {
            const int e = t + i * 128;              // 256 pairs of A elems
            const int m = e >> 4, kk2 = (e & 15) << 1;
            ta[i] = *(const float2*)(Ap + (mbase + m) * la + kk2);
        }
        float2 tb[8];
        if (btrans) {
#pragma unroll
            for (int i = 0; i < 8; ++i) {
                const int e = t + i * 128;          // pair along k
                const int n = e >> 4, kk2 = (e & 15) << 1;
                tb[i] = *(const float2*)(Bp + (nbase + n) * lb + koff + kk2);
            }
        } else {
#pragma unroll
            for (int i = 0; i < 8; ++i) {
                const int e = t + i * 128;          // pair along n
                const int kk = e >> 5, n2 = (e & 31) << 1;
                tb[i] = *(const float2*)(Bp + (koff + kk) * lb + nbase + n2);
            }
        }

        __syncthreads();   // previous iteration's operand reads complete
        // ---- phase 2: convert + LDS stores ----
#pragma unroll
        for (int i = 0; i < 2; ++i) {
            const int e = t + i * 128;
            const int m = e >> 4, kk2 = (e & 15) << 1;
            v2h p = { (_Float16)ta[i].x, (_Float16)ta[i].y };
            *(v2h*)(As + m * 32 + kk2) = p;          // ds_store_b32
        }
        if (btrans) {
#pragma unroll
            for (int i = 0; i < 8; ++i) {
                const int e = t + i * 128;
                const int n = e >> 4, kk2 = (e & 15) << 1;
                v2h p = { (_Float16)tb[i].x, (_Float16)tb[i].y };
                *(v2h*)(Bs + n * 32 + kk2) = p;      // ds_store_b32
            }
        } else {
#pragma unroll
            for (int i = 0; i < 8; ++i) {
                const int e = t + i * 128;
                const int kk = e >> 5, n2 = (e & 31) << 1;
                Bs[n2 * 32 + kk]       = (_Float16)tb[i].x;
                Bs[(n2 + 1) * 32 + kk] = (_Float16)tb[i].y;
            }
        }
        __syncthreads();

        // ---- operands: two ds_load_b128 each, compile-time concat ----
        const _Float16* arow = As + mn * 32 + hi * 8;
        const _Float16* brow = Bs + (wave * 16 + mn) * 32 + hi * 8;
        v8h a0 = *(const v8h*)(arow);
        v8h a1 = *(const v8h*)(arow + 16);
        v8h b0 = *(const v8h*)(brow);
        v8h b1 = *(const v8h*)(brow + 16);
        v16h a = __builtin_shufflevector(a0, a1, 0,1,2,3,4,5,6,7,8,9,10,11,12,13,14,15);
        v16h b = __builtin_shufflevector(b0, b1, 0,1,2,3,4,5,6,7,8,9,10,11,12,13,14,15);
        acc = __builtin_amdgcn_wmma_f32_16x16x32_f16(false, a, false, b,
                                                     (short)0, acc, false, false);
    }

    const int col = nbase + wave * 16 + mn;
    float bs = bias0 ? bias0[col] : 0.0f;
    if (bias1) bs += bias1[col];
#pragma unroll
    for (int rr = 0; rr < 8; ++rr) {
        const int row = mbase + rr + hi * 8;   // C/D layout: VGPR rr -> M=rr / 8+rr
        float v = acc[rr] + bs;
        if (act == 1) v = fmaxf(v, 0.0f);
        C[row * ldc + col] = v;
    }
}

// =====================================================================
// LSTM cell pointwise update (one step over the B axis; "batch" = L rows).
// gates[LPAD,512] order: i | f | g | o  (jnp.split order); rows >= LL ignored.
// =====================================================================
__global__ void lstm_pointwise_kernel(const float* __restrict__ gates,
                                      float* __restrict__ h, float* __restrict__ cs,
                                      float* __restrict__ outh)
{
    const int idx = blockIdx.x * blockDim.x + threadIdx.x;
    if (idx >= LL * DD) return;
    const int row = idx / DD, d = idx % DD;
    const float* g = gates + row * 4 * DD;
    const float gi = sigf(g[d]);
    const float gf = sigf(g[DD + d]);
    const float gg = tanhf(g[2 * DD + d]);
    const float go = sigf(g[3 * DD + d]);
    const float cn = gf * cs[idx] + gi * gg;
    const float hn = go * tanhf(cn);
    cs[idx] = cn;
    h[idx] = hn;
    outh[idx] = hn;
}

__global__ void fill_zero_kernel(float* __restrict__ p, int n)
{
    const int i = blockIdx.x * blockDim.x + threadIdx.x;
    if (i < n) p[i] = 0.0f;
}

// =====================================================================
// Final layer: y[n] = sigmoid([x1|x2][n,256] @ W_fc3[256,1] + b_fc3)
// =====================================================================
__global__ void fc3_sigmoid_kernel(const float* __restrict__ x1, const float* __restrict__ x2,
                                   const float* __restrict__ w, const float* __restrict__ b,
                                   float* __restrict__ out)
{
    const int n = blockIdx.x;
    const int t = threadIdx.x;            // 0..255
    const int lane = t & 31, wave = t >> 5;
    __shared__ float red[8];
    const float xv = (t < DD) ? x1[n * DD + t] : x2[n * DD + (t - DD)];
    float p = xv * w[t];
    for (int o = 16; o > 0; o >>= 1) p += __shfl_xor(p, o);
    if (lane == 0) red[wave] = p;
    __syncthreads();
    if (t == 0) {
        float s = b[0];
        for (int i = 0; i < 8; ++i) s += red[i];
        out[n] = 1.0f / (1.0f + expf(-s));
    }
}

// =====================================================================
// Host launcher
// =====================================================================
extern "C" void kernel_launch(void* const* d_in, const int* in_sizes, int n_in,
                              void* d_out, int out_size, void* d_ws, size_t ws_size,
                              hipStream_t stream)
{
    (void)in_sizes; (void)n_in; (void)out_size; (void)ws_size;

    // ---- inputs (setup_inputs dict order) ----
    const int*   q         = (const int*)  d_in[0];
    const int*   c         = (const int*)  d_in[1];
    const int*   shift_q   = (const int*)  d_in[2];
    const int*   shift_c   = (const int*)  d_in[3];
    const int*   r         = (const int*)  d_in[4];
    const int*   rgaps     = (const int*)  d_in[5];
    const int*   pcounts   = (const int*)  d_in[6];
    const int*   acounts   = (const int*)  d_in[7];
    const int*   s_rgaps   = (const int*)  d_in[8];
    const int*   s_pcounts = (const int*)  d_in[9];
    const int*   s_acounts = (const int*)  d_in[10];
    const float* E_skill   = (const float*)d_in[11];
    const float* E_question= (const float*)d_in[12];
    const float* E_response= (const float*)d_in[13];
    const float* wih_in    = (const float*)d_in[14];
    const float* whh_in    = (const float*)d_in[15];
    const float* bih_in    = (const float*)d_in[16];
    const float* bhh_in    = (const float*)d_in[17];
    const float* wih_fg    = (const float*)d_in[18];
    const float* whh_fg    = (const float*)d_in[19];
    const float* bih_fg    = (const float*)d_in[20];
    const float* bhh_fg    = (const float*)d_in[21];
    const float* W_l2      = (const float*)d_in[22];
    const float* b_l2      = (const float*)d_in[23];
    const float* W_l3      = (const float*)d_in[24];
    const float* b_l3      = (const float*)d_in[25];
    const float* W_fc3     = (const float*)d_in[26];
    const float* b_fc3     = (const float*)d_in[27];
    const float* W_fc4     = (const float*)d_in[28];
    const float* b_fc4     = (const float*)d_in[29];
    const float* W_fc5     = (const float*)d_in[30];
    const float* b_fc5     = (const float*)d_in[31];
    const float* W_cemb    = (const float*)d_in[32];
    const float* W_pre     = (const float*)d_in[33];
    const float* b_pre     = (const float*)d_in[34];
    float* out = (float*)d_out;

    // ---- workspace carve (floats); A-source buffers for the LSTM GEMM get
    //      16 padding rows so the M=208 unguarded tile loads stay in-bounds.
    float* ws = (float*)d_ws;
    float* xcat     = ws; ws += (size_t)NROW * 384;
    float* scat     = ws; ws += (size_t)NROW * 256;
    float* xemb     = ws; ws += (size_t)(NROW + 16) * DD;   // padded
    float* fh_in    = ws; ws += (size_t)(NROW + 16) * DD;   // padded
    float* fh_sh    = ws; ws += (size_t)NROW * DD;
    float* input_h  = ws; ws += (size_t)NROW * DD;
    float* forget_h = ws; ws += (size_t)NROW * DD;
    float* shift2   = ws; ws += (size_t)NROW * DD;
    float* x1       = ws; ws += (size_t)NROW * DD;
    float* x2       = ws; ws += (size_t)NROW * DD;
    float* hbuf     = ws; ws += (size_t)LPAD * DD;          // padded, zeroed
    float* csbuf    = ws; ws += (size_t)LL * DD;
    float* gates    = ws; ws += (size_t)LPAD * 4 * DD;      // padded

    // zero the padding rows of the LSTM A-sources (deterministic each call)
    fill_zero_kernel<<<(16 * DD + 255) / 256, 256, 0, stream>>>(xemb + NROW * DD, 16 * DD);
    fill_zero_kernel<<<(16 * DD + 255) / 256, 256, 0, stream>>>(fh_in + NROW * DD, 16 * DD);

    // Stage 1: attention + concat builders
    attn_concat_kernel<<<NROW, 128, 0, stream>>>(q, c, shift_q, shift_c, r,
                                                 E_question, E_skill, E_response,
                                                 xcat, scat);

    // Stage 2: sparse forget features (both variants)
    forget_kernel<<<NROW, 128, 0, stream>>>(c, rgaps, pcounts, acounts,
                                            E_skill, W_cemb, W_pre, b_pre, fh_in);
    forget_kernel<<<NROW, 128, 0, stream>>>(shift_c, s_rgaps, s_pcounts, s_acounts,
                                            E_skill, W_cemb, W_pre, b_pre, fh_sh);

    // Stage 3: xemb = xcat @ W_l2 + b_l2   (1600x384 * 384x128)
    gemm_wmma_kernel<<<dim3(NROW / 16, 128 / 64), 128, 0, stream>>>(
        xcat, xcat, 384, 384, 384,
        W_l2, W_l2, DD, DD, 0,
        b_l2, nullptr, xemb, DD, 384, 0);

    // Stage 4: shift2 = scat @ W_l3 + b_l3   (1600x256 * 256x128)
    gemm_wmma_kernel<<<dim3(NROW / 16, 128 / 64), 128, 0, stream>>>(
        scat, scat, 256, 256, 256,
        W_l3, W_l3, DD, DD, 0,
        b_l3, nullptr, shift2, DD, 256, 0);

    // Stage 5: two LSTMs (recurrence over axis0 = B = 8 steps, batch = L = 200)
    const float* xs[2]   = { xemb,    fh_in    };
    float*       outs[2] = { input_h, forget_h };
    const float* Wih[2]  = { wih_in, wih_fg };
    const float* Whh[2]  = { whh_in, whh_fg };
    const float* Bih[2]  = { bih_in, bih_fg };
    const float* Bhh[2]  = { bhh_in, bhh_fg };
    for (int li = 0; li < 2; ++li) {
        // zero h (LPAD rows; GEMM reads all 208) and cs (LL rows)
        fill_zero_kernel<<<((LPAD + LL) * DD + 255) / 256, 256, 0, stream>>>(
            hbuf, (LPAD + LL) * DD);
        for (int step = 0; step < BB; ++step) {
            // gates[208,512] = [x_step | h] @ [Wih^T ; Whh^T] + (bih + bhh)
            gemm_wmma_kernel<<<dim3(LPAD / 16, 512 / 64), 128, 0, stream>>>(
                xs[li] + step * LL * DD, hbuf, DD, DD, DD,
                Wih[li], Whh[li], DD, DD, /*btrans=*/1,
                Bih[li], Bhh[li], gates, 4 * DD, 2 * DD, 0);
            lstm_pointwise_kernel<<<(LL * DD + 255) / 256, 256, 0, stream>>>(
                gates, hbuf, csbuf, outs[li] + step * LL * DD);
        }
    }

    // Stage 6: x1 = relu([shift2 | input_h] @ W_fc4 + b_fc4)
    gemm_wmma_kernel<<<dim3(NROW / 16, 128 / 64), 128, 0, stream>>>(
        shift2, input_h, DD, DD, DD,
        W_fc4, W_fc4 + DD * DD, DD, DD, 0,
        b_fc4, nullptr, x1, DD, 2 * DD, 1);

    // Stage 7: x2 = relu([fh_shift | forget_h] @ W_fc5 + b_fc5)
    gemm_wmma_kernel<<<dim3(NROW / 16, 128 / 64), 128, 0, stream>>>(
        fh_sh, forget_h, DD, DD, DD,
        W_fc5, W_fc5 + DD * DD, DD, DD, 0,
        b_fc5, nullptr, x2, DD, 2 * DD, 1);

    // Stage 8: y = sigmoid([x1 | x2] @ W_fc3 + b_fc3)
    fc3_sigmoid_kernel<<<NROW, 256, 0, stream>>>(x1, x2, W_fc3, b_fc3, out);
}
